// ConcatCritic_13477607375614
// MI455X (gfx1250) — matrix operand order repaired and checked by reading, not compiled, not run on hardware
//
#include <hip/hip_runtime.h>
#include <hip/hip_bf16.h>

typedef __attribute__((ext_vector_type(16))) _Float16 v16h;
typedef __attribute__((ext_vector_type(8)))  float    v8f;
typedef __attribute__((ext_vector_type(4)))  float    v4f;

#define BATCH 1024
#define DIM   128

// A-matrix (16xK f16) per-lane K index for element e (0..15), lane-half hi:
// contiguous order, K = e + 16*hi (matches B fills)
__device__ __forceinline__ int KA(int e, int hi) {
    return e + 16 * hi;
}

// ---------------------------------------------------------------------------
// Stage 1: xa[a][k] = x[a] . W1[:128][:,k] + b1[k] ;  yb[b][k] = y[b] . W1[128:][:,k]
// ---------------------------------------------------------------------------
__global__ __launch_bounds__(128) void cc_proj_kernel(
    const float* __restrict__ x, const float* __restrict__ y,
    const float* __restrict__ W1, const float* __restrict__ b1,
    float* __restrict__ xa, float* __restrict__ yb) {
    __shared__ float srow[DIM];
    const int row = blockIdx.x;          // 0..2047 (first 1024 = x rows, rest = y rows)
    const int k   = threadIdx.x;         // 0..127
    const bool isX = (row < BATCH);
    const float* src = isX ? (x + row * DIM) : (y + (row - BATCH) * DIM);
    srow[k] = src[k];
    __syncthreads();
    const float* w = W1 + (isX ? 0 : DIM * DIM);   // W1[j][k] row-major, fan_in 256
    float acc = isX ? b1[k] : 0.0f;                // fold b1 into xa
    #pragma unroll 8
    for (int j = 0; j < DIM; ++j)
        acc = fmaf(srow[j], w[j * DIM + k], acc);
    float* dst = isX ? (xa + row * DIM) : (yb + (row - BATCH) * DIM);
    dst[k] = acc;
}

// ---------------------------------------------------------------------------
// Stage 2: per wave: 16 pair-columns (b), loop 16 rows (a).
// Transposed dataflow: h2^T = W2^T @ h1^T ; h3^T = W3^T(pad16) @ h2^T.
// ---------------------------------------------------------------------------
__global__ __launch_bounds__(256, 1) void cc_main_kernel(
    const float* __restrict__ xa, const float* __restrict__ yb,
    const float* __restrict__ W2, const float* __restrict__ b2,
    const float* __restrict__ W3, const float* __restrict__ b3,
    const float* __restrict__ W4, const float* __restrict__ b4p,
    float* __restrict__ out) {

    const int lane = threadIdx.x & 31;
    const int wave = threadIdx.x >> 5;
    const int l16  = lane & 15;
    const int hi   = lane >> 4;
    const int a0   = blockIdx.x * 16;
    const int b    = blockIdx.y * 128 + wave * 16 + l16;   // this lane's pair column

    // ---- preload W2^T as A-fragments: 4 M-tiles (feature m) x 4 K-steps ----
    v16h w2f[4][4];
    #pragma unroll
    for (int mt = 0; mt < 4; ++mt)
        #pragma unroll
        for (int ks = 0; ks < 4; ++ks)
            #pragma unroll
            for (int e = 0; e < 16; ++e) {
                int k = 32 * ks + KA(e, hi);               // h1 feature (K dim)
                w2f[mt][ks][e] = (_Float16)W2[k * 64 + 16 * mt + l16];
            }

    // ---- W3^T zero-padded to 16 rows, 2 K-steps ----
    v16h w3f[2];
    #pragma unroll
    for (int ks = 0; ks < 2; ++ks)
        #pragma unroll
        for (int e = 0; e < 16; ++e) {
            int k = 32 * ks + KA(e, hi);                   // h2 feature (K dim)
            w3f[ks][e] = (l16 < 8) ? (_Float16)W3[k * 8 + l16] : (_Float16)0.0f;
        }

    // ---- biases / head weights in registers ----
    float b2r[4][8];
    #pragma unroll
    for (int t = 0; t < 4; ++t)
        #pragma unroll
        for (int r = 0; r < 8; ++r)
            b2r[t][r] = b2[16 * t + r + 8 * hi];           // feature of C[t] reg r on this lane
    float b3r[8], w4r[8];
    #pragma unroll
    for (int r = 0; r < 8; ++r) { b3r[r] = b3[r]; w4r[r] = W4[r]; }
    const float b4s = b4p[0];

    // ---- preload yb B-fragment source values: contiguous 16-float windows ----
    float ybv[4][16];
    #pragma unroll
    for (int ks = 0; ks < 4; ++ks) {
        const v4f* p = (const v4f*)(yb + b * DIM + 32 * ks + 16 * hi);
        #pragma unroll
        for (int q = 0; q < 4; ++q) {
            v4f t = p[q];
            #pragma unroll
            for (int j = 0; j < 4; ++j) ybv[ks][4 * q + j] = t[j];
        }
    }

    for (int ai = 0; ai < 16; ++ai) {
        const int a = a0 + ai;
        const float* xrow = xa + a * DIM;
        __builtin_prefetch(xrow + DIM, 0, 3);              // global_prefetch next a row

        // ------------------ layer 2: h2^T = W2^T @ h1^T ------------------
        v8f c[4];
        #pragma unroll
        for (int mt = 0; mt < 4; ++mt) c[mt] = (v8f){0.f,0.f,0.f,0.f,0.f,0.f,0.f,0.f};

        #pragma unroll
        for (int ks = 0; ks < 4; ++ks) {
            const v4f* xp = (const v4f*)(xrow + 32 * ks + 16 * hi);
            union { v16h h; } bf;
            #pragma unroll
            for (int q = 0; q < 4; ++q) {
                v4f t = xp[q];
                #pragma unroll
                for (int j = 0; j < 4; ++j)
                    bf.h[4 * q + j] = (_Float16)fmaxf(t[j] + ybv[ks][4 * q + j], 0.0f);
            }
            #pragma unroll
            for (int mt = 0; mt < 4; ++mt)
                c[mt] = __builtin_amdgcn_wmma_f32_16x16x32_f16(
                    false, w2f[mt][ks], false, bf.h, (short)0, c[mt], false, false);
        }

        // ---- relu(h2 + b2) -> f16, packed as half2 dwords per C tile ----
        unsigned hv0[4], hv1[4], hv2[4], hv3[4];
        #pragma unroll
        for (int j = 0; j < 4; ++j) {
            #define PACK(dst, t)                                                       \
                {                                                                      \
                    _Float16 p0 = (_Float16)fmaxf(c[t][2*j]   + b2r[t][2*j],   0.0f);  \
                    _Float16 p1 = (_Float16)fmaxf(c[t][2*j+1] + b2r[t][2*j+1], 0.0f);  \
                    unsigned short u0 = __builtin_bit_cast(unsigned short, p0);        \
                    unsigned short u1 = __builtin_bit_cast(unsigned short, p1);        \
                    dst[j] = (unsigned)u0 | ((unsigned)u1 << 16);                      \
                }
            PACK(hv0, 0) PACK(hv1, 1) PACK(hv2, 2) PACK(hv3, 3)
            #undef PACK
        }

        // ------------------ layer 3: h3^T = W3^T @ h2^T ------------------
        // C layout already has lane==pair; only a xor-16 half-wave exchange is
        // needed to reach B layout: ds_swizzle SWAPX16 (xor=0x10, and=0x1f).
        v8f c3 = (v8f){0.f,0.f,0.f,0.f,0.f,0.f,0.f,0.f};
        #pragma unroll
        for (int ks = 0; ks < 2; ++ks) {
            union { v16h h; unsigned u[8]; } bf3;
            #pragma unroll
            for (int j = 0; j < 4; ++j) {
                unsigned e0 = (ks == 0) ? hv0[j] : hv2[j];   // tile 2*ks
                unsigned e1 = (ks == 0) ? hv1[j] : hv3[j];   // tile 2*ks+1
                unsigned own  = hi ? e1 : e0;                // my needed tile
                unsigned send = hi ? e0 : e1;                // partner's needed tile
                unsigned recv = (unsigned)__builtin_amdgcn_ds_swizzle((int)send, 0x401F);
                bf3.u[j]     = hi ? recv : own;              // elements 0..7  (K low)
                bf3.u[4 + j] = hi ? own  : recv;             // elements 8..15 (K high)
            }
            c3 = __builtin_amdgcn_wmma_f32_16x16x32_f16(
                false, w3f[ks], false, bf3.h, (short)0, c3, false, false);
        }

        // ---------- layer 4: in-lane relu(h3+b3) dot W4 ----------
        float acc = b4s;
        #pragma unroll
        for (int r = 0; r < 8; ++r)
            acc += fmaxf(c3[r] + b3r[r], 0.0f) * w4r[r];
        if (hi == 0)                                         // lanes 0..15 hold j=0..7 rows
            out[a * BATCH + b] = acc;
    }
}

// ---------------------------------------------------------------------------
extern "C" void kernel_launch(void* const* d_in, const int* in_sizes, int n_in,
                              void* d_out, int out_size, void* d_ws, size_t ws_size,
                              hipStream_t stream) {
    (void)in_sizes; (void)n_in; (void)out_size; (void)ws_size;
    const float* x  = (const float*)d_in[0];
    const float* y  = (const float*)d_in[1];
    const float* W1 = (const float*)d_in[2];
    const float* b1 = (const float*)d_in[3];
    const float* W2 = (const float*)d_in[4];
    const float* b2 = (const float*)d_in[5];
    const float* W3 = (const float*)d_in[6];
    const float* b3 = (const float*)d_in[7];
    const float* W4 = (const float*)d_in[8];
    const float* b4 = (const float*)d_in[9];
    float* out = (float*)d_out;

    float* xa = (float*)d_ws;                 // [1024][128] f32 (b1 folded in)
    float* yb = xa + BATCH * DIM;             // [1024][128] f32

    cc_proj_kernel<<<dim3(2 * BATCH), dim3(DIM), 0, stream>>>(x, y, W1, b1, xa, yb);
    cc_main_kernel<<<dim3(BATCH / 16, BATCH / 128), dim3(256), 0, stream>>>(
        xa, yb, W2, b2, W3, b3, W4, b4, out);
}
